// ResidualCrossAttention1_32908039422518
// MI455X (gfx1250) — compile-verified, matrix operand order checked by reading, not run on hardware
//
#include <hip/hip_runtime.h>

typedef __attribute__((ext_vector_type(16))) _Float16 v16h;
typedef __attribute__((ext_vector_type(8)))  float    v8f;

// ---- gfx1250 async Global->LDS path (guarded; falls back cleanly) ----------
#if defined(__has_builtin)
#  if __has_builtin(__builtin_amdgcn_global_load_async_to_lds_b128) && \
      __has_builtin(__builtin_amdgcn_s_wait_asynccnt)
#    define USE_ASYNC_LDS 1
#  endif
#endif
#ifndef USE_ASYNC_LDS
#  define USE_ASYNC_LDS 0
#endif

#if USE_ASYNC_LDS
// builtin signature (from clang diagnostic): (v4i AS1*, v4i AS3*, imm i32, imm i32)
typedef int v4i_ __attribute__((vector_size(4 * sizeof(int))));
typedef __attribute__((address_space(1))) v4i_* gv4i_p;   // global (printed __device__)
typedef __attribute__((address_space(3))) v4i_* lv4i_p;   // LDS    (printed __shared__)
#endif

namespace {
constexpr int kB  = 8;
constexpr int kC  = 128;
constexpr int kH  = 128;
constexpr int kW  = 128;
constexpr int kHW = kH * kW;
constexpr int kTP = 128;             // pixels per tile (kernels 3 & 7)
constexpr int kNT = kHW / kTP;       // 128 tiles per batch image

struct WeightPtrs {
  const float* w[6];    // 1x1 conv weights (C,C):  q1,k1,v1,q2,k2,v2
  const float* wb[6];   // 1x1 conv biases
  const float* dw[6];   // depthwise (C,1,3,3)
  const float* dwb[6];  // depthwise biases
};
} // namespace

// ---------------------------------------------------------------------------
// 1) convert GEMM weights to f16 once (A-fragments then stream from L2)
// ---------------------------------------------------------------------------
__global__ __launch_bounds__(256) void k_prep_weights(WeightPtrs wp, const float* __restrict__ lw,
                                                      _Float16* __restrict__ wf16,
                                                      _Float16* __restrict__ lwf16) {
  int idx = blockIdx.x * 256 + threadIdx.x;          // 131072 total
  if (idx < 6 * kC * kC) {
    int t = idx / (kC * kC);
    int r = idx % (kC * kC);
    wf16[idx] = (_Float16)wp.w[t][r];
  } else {
    int i  = idx - 6 * kC * kC;                      // lw is (C, 2C) row-major
    int o  = i / (2 * kC);
    int c2 = i % (2 * kC);
    int m  = c2 / kC, c = c2 % kC;                   // split into lw1|lw2
    lwf16[(size_t)m * kC * kC + o * kC + c] = (_Float16)lw[i];
  }
}

__global__ __launch_bounds__(256) void k_zero(float* __restrict__ p, int n) {
  int i = blockIdx.x * 256 + threadIdx.x;
  if (i < n) p[i] = 0.0f;
}

// ---------------------------------------------------------------------------
// 3) fused channel-LayerNorm + six 1x1 convs (WMMA GEMM over channels)
//    out[t][b][o][n] (f16), tile = 128 contiguous pixels of one batch image
// ---------------------------------------------------------------------------
__global__ __launch_bounds__(256) void k_ln_conv1x1(
    const float* __restrict__ Fi, const float* __restrict__ Fw,
    const float* __restrict__ g1, const float* __restrict__ g2,
    WeightPtrs wp, const _Float16* __restrict__ wf16,
    _Float16* __restrict__ conv1) {
  __shared__ __align__(32) _Float16 sX[2][kTP * kC];   // [fi/fw][p*128 + c] = 64 KB

  const int tid = threadIdx.x;
  const int b   = blockIdx.x >> 7;
  const int p0  = (blockIdx.x & 127) * kTP;

  // ---- per-pixel LayerNorm over the 128 channels (both inputs in parallel) ----
  {
    const int p   = tid & 127;
    const int sel = tid >> 7;
    const float* F = sel ? Fw : Fi;
    const float* g = sel ? g2 : g1;
    const size_t base = (size_t)b * kC * kHW + p0 + p;
    float s = 0.f, ss = 0.f;
    for (int c = 0; c < kC; ++c) {
      float v = F[base + (size_t)c * kHW];
      s += v; ss += v * v;
    }
    const float mean = s * (1.0f / kC);
    const float var  = ss * (1.0f / kC) - mean * mean;
    const float inv  = rsqrtf(var + 1e-5f);
    for (int c = 0; c < kC; ++c) {
      float v = F[base + (size_t)c * kHW];
      sX[sel][p * kC + c] = (_Float16)((v - mean) * inv * g[c]);
    }
  }
  __syncthreads();

  // ---- 6 GEMMs: out[o,p] = sum_c W[o,c] * X[p,c]; 384 16x16 tiles over 8 waves ----
  const int wave = tid >> 5, lane = tid & 31;
  const int row  = lane & 15, half = lane >> 4;
  for (int job = wave; job < 6 * 64; job += 8) {
    const int t  = job >> 6;
    const int ot = (job >> 3) & 7;
    const int pt = job & 7;
    const int s  = t / 3;                    // 0..2 use fi, 3..5 use fw
    v8f acc = {0.f,0.f,0.f,0.f,0.f,0.f,0.f,0.f};
    const _Float16* wb = wf16 + (size_t)t * kC * kC;
    for (int k = 0; k < kC; k += 32) {
      v16h a  = *(const v16h*)(wb + (ot * 16 + row) * kC + k + half * 16);
      v16h bf = *(const v16h*)(&sX[s][(pt * 16 + row) * kC + k + half * 16]);
      acc = __builtin_amdgcn_wmma_f32_16x16x32_f16(false, a, false, bf, (short)0, acc, false, false);
    }
    const float* bias = wp.wb[t];
    _Float16* obase = conv1 + (((size_t)t * kB + b) * kC) * kHW + p0;
    for (int j = 0; j < 8; ++j) {
      int o  = ot * 16 + j + 8 * half;       // C/D layout: VGPR j -> M = j + 8*(lane/16)
      int pp = pt * 16 + row;
      obase[(size_t)o * kHW + pp] = (_Float16)(acc[j] + bias[o]);
    }
  }
}

// ---------------------------------------------------------------------------
// 4) depthwise 3x3 SAME conv + bias, 32x32 tile with 34x34 halo in LDS
// ---------------------------------------------------------------------------
__global__ __launch_bounds__(256) void k_dwconv3x3(
    WeightPtrs wp, const _Float16* __restrict__ conv1, _Float16* __restrict__ qkv) {
  __shared__ _Float16 sT[34 * 34];
  const int tid   = threadIdx.x;
  const int tile  = blockIdx.x & 15;
  const int plane = blockIdx.x >> 4;
  const int t = plane / (kB * kC);
  const int r = plane % (kB * kC);
  const int b = r / kC;
  const int c = r % kC;
  const int oy0 = (tile >> 2) * 32;
  const int ox0 = (tile & 3) * 32;

  const _Float16* in = conv1 + (((size_t)t * kB + b) * kC + c) * kHW;
  for (int i = tid; i < 34 * 34; i += 256) {
    int yy = i / 34 - 1 + oy0;
    int xx = i % 34 - 1 + ox0;
    _Float16 v = (_Float16)0.f;
    if (yy >= 0 && yy < kH && xx >= 0 && xx < kW) v = in[yy * kW + xx];
    sT[i] = v;
  }
  __syncthreads();

  float wr[9];
  const float* dwp = wp.dw[t] + c * 9;
  for (int i = 0; i < 9; ++i) wr[i] = dwp[i];
  const float bv = wp.dwb[t][c];
  _Float16* out = qkv + (((size_t)t * kB + b) * kC + c) * kHW;
  for (int i = tid; i < 32 * 32; i += 256) {
    int ly = i >> 5, lx = i & 31;
    float acc = bv;
    for (int ky = 0; ky < 3; ++ky)
      for (int kx = 0; kx < 3; ++kx)
        acc += wr[ky * 3 + kx] * (float)sT[(ly + ky) * 34 + (lx + kx)];
    out[(oy0 + ly) * kW + (ox0 + lx)] = (_Float16)acc;
  }
}

// ---------------------------------------------------------------------------
// 5) attention logits: S[c,d] = sum_n Q[c,n]*K[d,n], split-K x8 + f32 atomics
// ---------------------------------------------------------------------------
__global__ __launch_bounds__(256) void k_attn_logits(
    const _Float16* __restrict__ qkv, float* __restrict__ logits) {
  const int tid  = threadIdx.x;
  const int wave = tid >> 5, lane = tid & 31;
  const int row  = lane & 15, half = lane >> 4;
  const int job  = blockIdx.x * 8 + wave;   // 8192 wave-jobs
  const int ks   = job & 7;
  const int dt   = (job >> 3) & 7;
  const int ct   = (job >> 6) & 7;
  const int b    = (job >> 9) & 7;
  const int pair = job >> 12;               // 0: Qi.Kw (a1), 1: Qw.Ki (a2)

  const int tq = pair ? 3 : 0;
  const int tk = pair ? 1 : 4;
  const _Float16* Q = qkv + (((size_t)tq * kB + b) * kC) * kHW;
  const _Float16* K = qkv + (((size_t)tk * kB + b) * kC) * kHW;
  const int n0 = ks * (kHW / 8);

  v8f acc = {0.f,0.f,0.f,0.f,0.f,0.f,0.f,0.f};
  const _Float16* qrow = Q + (size_t)(ct * 16 + row) * kHW + half * 16;
  const _Float16* krow = K + (size_t)(dt * 16 + row) * kHW + half * 16;
  for (int n = n0; n < n0 + kHW / 8; n += 32) {
    // stream hint: pull the next cachelines while WMMA consumes these
    __builtin_prefetch(qrow + n + 256, 0, 1);   // -> global_prefetch_b8
    __builtin_prefetch(krow + n + 256, 0, 1);
    v16h a  = *(const v16h*)(qrow + n);
    v16h bf = *(const v16h*)(krow + n);
    acc = __builtin_amdgcn_wmma_f32_16x16x32_f16(false, a, false, bf, (short)0, acc, false, false);
  }
  float* lg = logits + (((size_t)pair * kB + b) * kC) * kC;
  for (int j = 0; j < 8; ++j) {
    int cc = ct * 16 + j + 8 * half;
    int dd = dt * 16 + row;
    atomicAdd(&lg[(size_t)cc * kC + dd], acc[j]);
  }
}

// ---------------------------------------------------------------------------
// 6) row softmax (128 wide), one wave32 per row, emit f16 attention
// ---------------------------------------------------------------------------
__global__ __launch_bounds__(256) void k_softmax(
    const float* __restrict__ logits, _Float16* __restrict__ attn) {
  const int wave = threadIdx.x >> 5, lane = threadIdx.x & 31;
  const int rowi = blockIdx.x * 8 + wave;    // 2048 rows
  const size_t base = (size_t)rowi * kC + lane * 4;
  float v[4];
  for (int i = 0; i < 4; ++i) v[i] = logits[base + i];
  float m = fmaxf(fmaxf(v[0], v[1]), fmaxf(v[2], v[3]));
  for (int off = 16; off > 0; off >>= 1) m = fmaxf(m, __shfl_xor(m, off, 32));
  float s = 0.f;
  for (int i = 0; i < 4; ++i) { v[i] = __expf(v[i] - m); s += v[i]; }
  for (int off = 16; off > 0; off >>= 1) s += __shfl_xor(s, off, 32);
  const float r = 1.0f / s;
  for (int i = 0; i < 4; ++i) attn[base + i] = (_Float16)(v[i] * r);
}

// ---------------------------------------------------------------------------
// 7) apply attention: A(128x128) @ V(128 x ptile), write Aw/Ai in [p][c] f16
//    V tiles staged in LDS; async Global->LDS DMA when the toolchain has it.
// ---------------------------------------------------------------------------
__global__ __launch_bounds__(256) void k_attn_apply(
    const _Float16* __restrict__ qkv, const _Float16* __restrict__ attn,
    _Float16* __restrict__ awbuf) {
  // async path: natural [m][d*128 + p] layout (contiguous 16B on both sides)
  // fallback:  transposed [m][p*128 + d] layout (vector B-fragment loads)
  __shared__ __align__(32) _Float16 sV[2][kTP * kC];   // 64 KB
  const int tid = threadIdx.x;
  const int b   = blockIdx.x >> 7;
  const int p0  = (blockIdx.x & 127) * kTP;

#if USE_ASYNC_LDS
  for (int ch = tid; ch < 2 * kTP * kC / 8; ch += 256) {   // 16 B per lane per issue
    const int e  = ch * 8;
    const int m  = e >= kTP * kC;            // 0: Vw, 1: Vi
    const int r2 = e & (kTP * kC - 1);
    const int d  = r2 >> 7;
    const int pp = r2 & 127;                 // multiple of 8
    const int t  = m ? 2 : 5;
    const _Float16* gp = &qkv[(((size_t)t * kB + b) * kC + d) * kHW + p0 + pp];
    _Float16* lp = &sV[m][d * kTP + pp];
    __builtin_amdgcn_global_load_async_to_lds_b128((gv4i_p)(void*)(const_cast<_Float16*>(gp)),
                                                   (lv4i_p)(void*)lp, 0, 0);
  }
  __builtin_amdgcn_s_wait_asynccnt(0);
#else
  for (int i = tid; i < 2 * kTP * kC; i += 256) {
    const int m  = i >= kTP * kC;            // 0: Vw, 1: Vi
    const int r2 = i & (kTP * kC - 1);
    const int d  = r2 >> 7;
    const int pp = r2 & 127;                 // consecutive i -> coalesced global read
    const int t  = m ? 2 : 5;
    sV[m][pp * kC + d] = qkv[(((size_t)t * kB + b) * kC + d) * kHW + p0 + pp];
  }
#endif
  __syncthreads();

  const int wave = tid >> 5, lane = tid & 31;
  const int row  = lane & 15, half = lane >> 4;
  for (int job = wave; job < 128; job += 8) {
    const int m  = job >> 6;
    const int ct = (job >> 3) & 7;
    const int pt = job & 7;
    const _Float16* A = attn + (((size_t)m * kB + b) * kC) * kC;  // a1/a2 [c][d]
    v8f acc = {0.f,0.f,0.f,0.f,0.f,0.f,0.f,0.f};
    for (int k = 0; k < kC; k += 32) {
      v16h a = *(const v16h*)(A + (ct * 16 + row) * kC + k + half * 16);
      v16h bf;
#if USE_ASYNC_LDS
      // gather the B-column (16 k-values for pixel column) from [d][p] layout
      #pragma unroll
      for (int kk = 0; kk < 16; ++kk)
        bf[kk] = sV[m][(k + half * 16 + kk) * kTP + pt * 16 + row];
#else
      bf = *(const v16h*)(&sV[m][(pt * 16 + row) * kC + k + half * 16]);
#endif
      acc = __builtin_amdgcn_wmma_f32_16x16x32_f16(false, a, false, bf, (short)0, acc, false, false);
    }
    _Float16* ob = awbuf + (((size_t)m * kB + b) * kHW + p0) * kC;
    for (int j = 0; j < 8; ++j) {
      int cc = ct * 16 + j + 8 * half;
      int pp = pt * 16 + row;
      ob[(size_t)pp * kC + cc] = (_Float16)acc[j];   // [p][c]: contiguous c for kernel 8
    }
  }
}

// ---------------------------------------------------------------------------
// 8) fused concat + linear (K=256) + bias + residual, f32 output
// ---------------------------------------------------------------------------
__global__ __launch_bounds__(256) void k_out_linear(
    const _Float16* __restrict__ lwf16, const float* __restrict__ lb,
    const _Float16* __restrict__ awbuf,
    const float* __restrict__ Fi, const float* __restrict__ Fw,
    float* __restrict__ out) {
  const int tid  = threadIdx.x;
  const int wave = tid >> 5, lane = tid & 31;
  const int row  = lane & 15, half = lane >> 4;
  const int job  = blockIdx.x * 8 + wave;    // 65536 wave-jobs
  const int ptile = job & 1023;
  const int ot    = (job >> 10) & 7;
  const int b     = job >> 13;
  const int p0    = ptile * 16;

  v8f acc = {0.f,0.f,0.f,0.f,0.f,0.f,0.f,0.f};
  for (int m = 0; m < 2; ++m) {
    const _Float16* L  = lwf16 + (size_t)m * kC * kC;                 // lw1 / lw2
    const _Float16* Bm = awbuf + (((size_t)m * kB + b) * kHW + p0) * kC;
    for (int k = 0; k < kC; k += 32) {
      v16h a  = *(const v16h*)(L + (ot * 16 + row) * kC + k + half * 16);
      v16h bf = *(const v16h*)(Bm + (size_t)row * kC + k + half * 16);
      acc = __builtin_amdgcn_wmma_f32_16x16x32_f16(false, a, false, bf, (short)0, acc, false, false);
    }
  }
  for (int j = 0; j < 8; ++j) {
    const int o  = ot * 16 + j + 8 * half;
    const int p  = p0 + row;
    const size_t idx = ((size_t)b * kC + o) * kHW + p;
    out[idx] = acc[j] + lb[o] + Fi[idx] + Fw[idx];
  }
}

// ---------------------------------------------------------------------------
extern "C" void kernel_launch(void* const* d_in, const int* in_sizes, int n_in,
                              void* d_out, int out_size, void* d_ws, size_t ws_size,
                              hipStream_t stream) {
  (void)in_sizes; (void)n_in; (void)out_size; (void)ws_size;
  const float* Fi = (const float*)d_in[0];
  const float* Fw = (const float*)d_in[1];
  const float* g1 = (const float*)d_in[2];
  const float* g2 = (const float*)d_in[3];
  WeightPtrs wp;
  for (int i = 0; i < 6; ++i) {
    wp.w[i]   = (const float*)d_in[4 + i];
    wp.wb[i]  = (const float*)d_in[10 + i];
    wp.dw[i]  = (const float*)d_in[16 + i];
    wp.dwb[i] = (const float*)d_in[22 + i];
  }
  const float* lw = (const float*)d_in[28];
  const float* lb = (const float*)d_in[29];
  float* out = (float*)d_out;

  char* ws = (char*)d_ws;
  size_t off = 0;
  _Float16* wf16   = (_Float16*)(ws + off); off += (size_t)6 * kC * kC * 2;
  _Float16* lwf16  = (_Float16*)(ws + off); off += (size_t)2 * kC * kC * 2;
  _Float16* conv1  = (_Float16*)(ws + off); off += (size_t)6 * kB * kC * kHW * 2;
  _Float16* qkv    = (_Float16*)(ws + off); off += (size_t)6 * kB * kC * kHW * 2;
  float*    logits = (float*)   (ws + off); off += (size_t)2 * kB * kC * kC * 4;
  _Float16* attn   = (_Float16*)(ws + off); off += (size_t)2 * kB * kC * kC * 2;
  _Float16* awbuf  = conv1;   // conv1 is dead after the depthwise stage -> reuse

  hipLaunchKernelGGL(k_prep_weights, dim3(512), dim3(256), 0, stream, wp, lw, wf16, lwf16);
  hipLaunchKernelGGL(k_zero,         dim3(1024), dim3(256), 0, stream, logits, 2 * kB * kC * kC);
  hipLaunchKernelGGL(k_ln_conv1x1,   dim3(kB * kNT), dim3(256), 0, stream, Fi, Fw, g1, g2, wp, wf16, conv1);
  hipLaunchKernelGGL(k_dwconv3x3,    dim3(6 * kB * kC * 16), dim3(256), 0, stream, wp, conv1, qkv);
  hipLaunchKernelGGL(k_attn_logits,  dim3(1024), dim3(256), 0, stream, qkv, logits);
  hipLaunchKernelGGL(k_softmax,      dim3(256),  dim3(256), 0, stream, logits, attn);
  hipLaunchKernelGGL(k_attn_apply,   dim3(kB * kNT), dim3(256), 0, stream, qkv, attn, awbuf);
  hipLaunchKernelGGL(k_out_linear,   dim3(8192), dim3(256), 0, stream, lwf16, lb, awbuf, Fi, Fw, out);
}
